// AlexandriaMultiheadAttention_78743930405047
// MI455X (gfx1250) — matrix operand order repaired and checked by reading, not compiled, not run on hardware
//
#include <hip/hip_runtime.h>

// Multihead causal attention, fp32 in/out, f16 WMMA compute (f32 accum).
//   B=4, H=16, S=2048, D=1024, Dh=64
// Pipeline: cvt(f32->f16) -> QKV projection GEMM (LDS-tiled WMMA, TDM-staged)
//           -> flash attention (TDM double-buffered K/V tiles, WMMA).

#define D_MODEL 1024
#define N_HEADS 16
#define SEQ     2048
#define BATCH   4
#define DH      64

typedef __attribute__((ext_vector_type(16))) _Float16 v16h;
typedef __attribute__((ext_vector_type(8)))  _Float16 v8h;
typedef __attribute__((ext_vector_type(8)))  float    v8f;
typedef __attribute__((ext_vector_type(4)))  unsigned int u32x4;
typedef __attribute__((ext_vector_type(8)))  int      i32x8;
typedef __attribute__((ext_vector_type(4)))  int      i32x4;

union Frag { v16h v; v8h h[2]; };

#if defined(__has_builtin)
#if __has_builtin(__builtin_amdgcn_tensor_load_to_lds) && \
    __has_builtin(__builtin_amdgcn_s_wait_tensorcnt)
#define USE_TDM 1
#endif
#endif
#ifndef USE_TDM
#define USE_TDM 0
#endif

static __device__ __forceinline__ v8f wmma_f16(v16h a, v16h b, v8f c) {
  // D = A(16x32 f16) * B(32x16 f16) + C(16x16 f32)
  return __builtin_amdgcn_wmma_f32_16x16x32_f16(false, a, false, b, (short)0, c,
                                                false, false);
}

#if USE_TDM
// 2-D tile DMA: global (tile1 rows x tile0 elems of f16, row stride stride0)
// -> LDS at lds_off, with pad_amount DWORDs of LDS padding inserted every
// pad_interval-coded DWORDs (CDNA5 ISA).
static __device__ __forceinline__ void tdm_load_2d(
    unsigned lds_off, const _Float16* gsrc,
    unsigned tile0, unsigned tile1, unsigned stride0,
    unsigned pad_interval_code, unsigned pad_amount_code) {
  unsigned long long ga = (unsigned long long)gsrc;
  u32x4 g0;
  g0[0] = 1u;                                   // count=1, no gather
  g0[1] = lds_off;                              // LDS byte address
  g0[2] = (unsigned)ga;                         // global_addr[31:0]
  g0[3] = (unsigned)((ga >> 32) & 0x1ffffffull) | (2u << 30);  // [56:32]|type=2
  i32x8 g1;
  g1[0] = (int)((1u << 16)                      // data_size = 2 bytes
              | (1u << 20)                      // pad_enable
              | (pad_interval_code << 22) | (pad_amount_code << 25));
  g1[1] = (int)(tile0 << 16);                   // tensor_dim0[15:0] @ bits63:48
  g1[2] = (int)(tile1 << 16);                   // dim0 hi=0 | tensor_dim1[15:0]
  g1[3] = (int)(tile0 << 16);                   // dim1 hi=0 | tile_dim0
  g1[4] = (int)tile1;                           // tile_dim1 ; tile_dim2=0
  g1[5] = (int)stride0;                         // tensor_dim0_stride[31:0]
  g1[6] = 0; g1[7] = 0;
  i32x4 z4 = {0, 0, 0, 0};                      // groups 2/3 unused (2-D)
  i32x8 z8 = {0, 0, 0, 0, 0, 0, 0, 0};          // 6-arg form: extra group
  __builtin_amdgcn_tensor_load_to_lds(g0, g1, z4, z4, z8, 0);
}
#endif

// ---------------------------------------------------------------------------
// f32 -> f16 conversion (8 elements / thread), optional scale fold-in.
// ---------------------------------------------------------------------------
__global__ void cvt_kernel(const float* __restrict__ src, _Float16* __restrict__ dst,
                           int n8, float scale) {
  int i = blockIdx.x * blockDim.x + threadIdx.x;
  if (i >= n8) return;
  const float4* s = (const float4*)src + (size_t)i * 2;
  float4 f0 = s[0], f1 = s[1];
  v8h o;
  o[0] = (_Float16)(f0.x * scale); o[1] = (_Float16)(f0.y * scale);
  o[2] = (_Float16)(f0.z * scale); o[3] = (_Float16)(f0.w * scale);
  o[4] = (_Float16)(f1.x * scale); o[5] = (_Float16)(f1.y * scale);
  o[6] = (_Float16)(f1.z * scale); o[7] = (_Float16)(f1.w * scale);
  ((v8h*)dst)[i] = o;
}

// ---------------------------------------------------------------------------
// QKV projection: Y = xh @ Wh[sel]^T, M=8192, N=1024, K=1024.
// Block (8 waves) computes a 128x128 tile; A/B 128x32 panels staged in LDS
// (TDM double-buffered; padded rows -> conflict-free ds_load_b128 fragments).
// Each wave: 2 M-subtiles x 4 N-subtiles = 8 WMMA per K-step.
// Outputs in attention layouts:
//   sel 0 -> Qh[B][H][S][Dh] (pre-scaled 1/8 via Wq conversion)
//   sel 1 -> Kh[B][H][S][Dh]
//   sel 2 -> Vt[B][H][Dh][S]
// ---------------------------------------------------------------------------
__global__ __launch_bounds__(256) void proj_kernel(
    const _Float16* __restrict__ xh, const _Float16* __restrict__ wh,
    _Float16* __restrict__ qh, _Float16* __restrict__ kh, _Float16* __restrict__ vt) {
  __shared__ __align__(16) _Float16 Ab[2][128 * 40];  // 32+8 pad halfs per row
  __shared__ __align__(16) _Float16 Bb[2][128 * 40];

  const int bid = blockIdx.x;
  const int sel = bid >> 9;                 // 0:Q 1:K 2:V  (512 blocks each)
  const int rrr = bid & 511;
  const int m0  = (rrr & 63) * 128;         // 64 M-tiles
  const int n0  = (rrr >> 6) * 128;         // 8  N-tiles
  const int wid  = threadIdx.x >> 5;
  const int lane = threadIdx.x & 31;
  const int rlo = lane & 15, hi = lane >> 4;
  const int wm = (wid & 3) * 32;            // wave row base within tile
  const int wn = (wid >> 2) * 64;           // wave col base within tile

  const _Float16* w     = wh + (size_t)sel * (D_MODEL * D_MODEL);
  const _Float16* abase = xh + (size_t)m0 * D_MODEL;
  const _Float16* bbase = w  + (size_t)n0 * D_MODEL;

  v8f acc[2][4] = {};

#if USE_TDM
  const bool issuer = (wid == 0);
  if (issuer) {
    tdm_load_2d((unsigned)(unsigned long long)&Ab[0][0], abase, 32, 128, D_MODEL, 3, 3);
    tdm_load_2d((unsigned)(unsigned long long)&Bb[0][0], bbase, 32, 128, D_MODEL, 3, 3);
  }
#endif

  const int NS = D_MODEL / 32;
  for (int ks = 0; ks < NS; ++ks) {
    const int cur = USE_TDM ? (ks & 1) : 0;
#if USE_TDM
    if (issuer) {
      if (ks + 1 < NS) {  // overlap next panel DMA with this step's compute
        const int k1 = (ks + 1) * 32;
        tdm_load_2d((unsigned)(unsigned long long)&Ab[cur ^ 1][0], abase + k1,
                    32, 128, D_MODEL, 3, 3);
        tdm_load_2d((unsigned)(unsigned long long)&Bb[cur ^ 1][0], bbase + k1,
                    32, 128, D_MODEL, 3, 3);
        __builtin_amdgcn_s_wait_tensorcnt((short)2);  // current panel ready
      } else {
        __builtin_amdgcn_s_wait_tensorcnt((short)0);
      }
    }
    __syncthreads();
#else
    {  // cooperative copy fallback (single buffer)
      const int k0 = ks * 32;
      const int row = threadIdx.x >> 1, c = (threadIdx.x & 1) * 16;
      *(v8h*)&Ab[0][row * 40 + c] = *(const v8h*)(abase + (size_t)row * D_MODEL + k0 + c);
      *(v8h*)&Bb[0][row * 40 + c] = *(const v8h*)(bbase + (size_t)row * D_MODEL + k0 + c);
    }
    __syncthreads();
#endif

    Frag a[2];
#pragma unroll
    for (int i = 0; i < 2; ++i) {
      const _Float16* p = &Ab[cur][(wm + i * 16 + rlo) * 40 + hi * 8];
      a[i].h[0] = *(const v8h*)p;
      a[i].h[1] = *(const v8h*)(p + 16);
    }
#pragma unroll
    for (int j = 0; j < 4; ++j) {
      const _Float16* p = &Bb[cur][(wn + j * 16 + rlo) * 40 + hi * 8];
      Frag bf;
      bf.h[0] = *(const v8h*)p;
      bf.h[1] = *(const v8h*)(p + 16);
#pragma unroll
      for (int i = 0; i < 2; ++i)
        acc[i][j] = wmma_f16(a[i].v, bf.v, acc[i][j]);
    }
    __syncthreads();  // all waves done reading before next DMA overwrites
  }

#pragma unroll
  for (int i = 0; i < 2; ++i)
#pragma unroll
    for (int j = 0; j < 4; ++j)
#pragma unroll
      for (int r = 0; r < 8; ++r) {
        const int row = m0 + wm + i * 16 + r + hi * 8;
        const int col = n0 + wn + j * 16 + rlo;
        const int b = row >> 11, s = row & (SEQ - 1);
        const int h = col >> 6,  d = col & (DH - 1);
        const size_t bhh = (size_t)(b * N_HEADS + h);
        const _Float16 val = (_Float16)acc[i][j][r];
        if (sel == 0)      qh[(bhh * SEQ + s) * DH + d] = val;
        else if (sel == 1) kh[(bhh * SEQ + s) * DH + d] = val;
        else               vt[(bhh * DH + d) * SEQ + s] = val;
      }
}

// ---------------------------------------------------------------------------
// Flash attention: one wave handles 16 q-rows of one (b,h); 32-key blocks
// streamed through TDM double-buffered LDS tiles (K 32x64 pad->72, V 64x32
// pad->40), online softmax, P repacked via 1KB LDS tile, WMMA for QK^T & PV.
// ---------------------------------------------------------------------------
__global__ __launch_bounds__(32) void attn_kernel(
    const _Float16* __restrict__ qh, const _Float16* __restrict__ kh,
    const _Float16* __restrict__ vt, float* __restrict__ out) {
  __shared__ __align__(16) _Float16 Kt[2][32 * 72];
  __shared__ __align__(16) _Float16 Vs[2][64 * 40];
  __shared__ __align__(16) _Float16 pl[16 * 32];

  const int qb = blockIdx.x;   // 0..127
  const int h  = blockIdx.y;
  const int b  = blockIdx.z;
  const int lane = threadIdx.x & 31;
  const int rlo = lane & 15, hi = lane >> 4;
  const size_t bh = (size_t)(b * N_HEADS + h);
  const _Float16* qbase = qh + bh * SEQ * DH;
  const _Float16* kbase = kh + bh * SEQ * DH;
  const _Float16* vbase = vt + bh * DH * SEQ;

  Frag qa[2];
#pragma unroll
  for (int c = 0; c < 2; ++c) {
    const _Float16* p = qbase + (size_t)(qb * 16 + rlo) * DH + c * 32 + hi * 8;
    qa[c].h[0] = *(const v8h*)p;
    qa[c].h[1] = *(const v8h*)(p + 16);
  }

  float m[8], l[8];
  v8f O[4] = {};
#pragma unroll
  for (int r = 0; r < 8; ++r) { m[r] = -3.0e38f; l[r] = 0.0f; }

  const int nkb = (qb * 16 + 15) / 32 + 1;

#if USE_TDM
  tdm_load_2d((unsigned)(unsigned long long)&Kt[0][0], kbase, 64, 32, DH, 4, 3);
  tdm_load_2d((unsigned)(unsigned long long)&Vs[0][0], vbase, 32, 64, SEQ, 3, 3);
#endif

  for (int kb = 0; kb < nkb; ++kb) {
    const int k0 = kb * 32;
    const int cur = USE_TDM ? (kb & 1) : 0;
#if USE_TDM
    // previous iteration's LDS reads must land before DMA reuses that buffer
    asm volatile("s_wait_dscnt 0x0" ::: "memory");
    if (kb + 1 < nkb) {
      tdm_load_2d((unsigned)(unsigned long long)&Kt[cur ^ 1][0],
                  kbase + (size_t)(k0 + 32) * DH, 64, 32, DH, 4, 3);
      tdm_load_2d((unsigned)(unsigned long long)&Vs[cur ^ 1][0],
                  vbase + k0 + 32, 32, 64, SEQ, 3, 3);
      __builtin_amdgcn_s_wait_tensorcnt((short)2);  // current tiles ready
    } else {
      __builtin_amdgcn_s_wait_tensorcnt((short)0);
    }
    asm volatile("" ::: "memory");
#else
    for (int u = lane; u < 128; u += 32) {  // K: 32 rows x 4 chunks
      const int row = u >> 2, c = (u & 3) * 16;
      *(v8h*)&Kt[0][row * 72 + c] = *(const v8h*)(kbase + (size_t)(k0 + row) * DH + c);
    }
    for (int u = lane; u < 128; u += 32) {  // V: 64 rows x 2 chunks
      const int row = u >> 1, c = (u & 1) * 16;
      *(v8h*)&Vs[0][row * 40 + c] = *(const v8h*)(vbase + (size_t)row * SEQ + k0 + c);
    }
    __syncthreads();
#endif

    // Scores from LDS K tile
    v8f st[2];
#pragma unroll
    for (int t = 0; t < 2; ++t) {
      st[t] = (v8f){};
#pragma unroll
      for (int c = 0; c < 2; ++c) {
        const _Float16* p = &Kt[cur][(t * 16 + rlo) * 72 + c * 32 + hi * 8];
        Frag kf;
        kf.h[0] = *(const v8h*)p;
        kf.h[1] = *(const v8h*)(p + 16);
        st[t] = wmma_f16(qa[c].v, kf.v, st[t]);
      }
    }

    // Causal mask needed only in the diagonal (last) block with this tiling.
    if (kb == nkb - 1) {
#pragma unroll
      for (int t = 0; t < 2; ++t)
#pragma unroll
        for (int r = 0; r < 8; ++r) {
          const int key = k0 + t * 16 + rlo;
          const int row = qb * 16 + r + hi * 8;
          if (key > row) st[t][r] = -3.0e38f;
        }
    }

    // Online softmax per row (rows across 16-lane halves in C layout).
#pragma unroll
    for (int r = 0; r < 8; ++r) {
      float mx = fmaxf(st[0][r], st[1][r]);
#pragma unroll
      for (int off = 1; off < 16; off <<= 1)
        mx = fmaxf(mx, __shfl_xor(mx, off, 32));
      const float mn = fmaxf(m[r], mx);
      const float sc = exp2f((m[r] - mn) * 1.44269504f);
      const float p0 = exp2f((st[0][r] - mn) * 1.44269504f);
      const float p1 = exp2f((st[1][r] - mn) * 1.44269504f);
      float sum = p0 + p1;
#pragma unroll
      for (int off = 1; off < 16; off <<= 1)
        sum += __shfl_xor(sum, off, 32);
      m[r] = mn;
      l[r] = l[r] * sc + sum;
#pragma unroll
      for (int d = 0; d < 4; ++d) O[d][r] *= sc;
      pl[(r + hi * 8) * 32 + rlo]      = (_Float16)p0;
      pl[(r + hi * 8) * 32 + 16 + rlo] = (_Float16)p1;
    }
    __syncthreads();

    Frag pa;  // P as A-fragment (16x32 keys)
    pa.h[0] = *(const v8h*)&pl[rlo * 32 + hi * 8];
    pa.h[1] = *(const v8h*)&pl[rlo * 32 + hi * 8 + 16];
    __syncthreads();

    // O += P @ V from LDS V tile
#pragma unroll
    for (int d = 0; d < 4; ++d) {
      const _Float16* p = &Vs[cur][(d * 16 + rlo) * 40 + hi * 8];
      Frag vf;
      vf.h[0] = *(const v8h*)p;
      vf.h[1] = *(const v8h*)(p + 16);
      O[d] = wmma_f16(pa.v, vf.v, O[d]);
    }
#if !USE_TDM
    __syncthreads();
#endif
  }

#pragma unroll
  for (int r = 0; r < 8; ++r) {
    const float inv = 1.0f / l[r];
    const int row = qb * 16 + r + hi * 8;
#pragma unroll
    for (int d = 0; d < 4; ++d)
      out[(bh * SEQ + row) * DH + d * 16 + rlo] = O[d][r] * inv;
  }
}

// ---------------------------------------------------------------------------
extern "C" void kernel_launch(void* const* d_in, const int* in_sizes, int n_in,
                              void* d_out, int out_size, void* d_ws, size_t ws_size,
                              hipStream_t stream) {
  (void)in_sizes; (void)n_in; (void)out_size; (void)ws_size;
  const float* x  = (const float*)d_in[0];
  const float* Wq = (const float*)d_in[1];
  const float* Wk = (const float*)d_in[2];
  const float* Wv = (const float*)d_in[3];
  float* out = (float*)d_out;

  // Workspace carve (halfs): xh | wh[3] | qh | kh | vt  (~73.4 MB total)
  _Float16* xh = (_Float16*)d_ws;
  _Float16* wh = xh + (size_t)BATCH * SEQ * D_MODEL;
  _Float16* qh = wh + (size_t)3 * D_MODEL * D_MODEL;
  _Float16* kh = qh + (size_t)BATCH * SEQ * D_MODEL;
  _Float16* vt = kh + (size_t)BATCH * SEQ * D_MODEL;

  const int nx8 = BATCH * SEQ * D_MODEL / 8;
  const int nw8 = D_MODEL * D_MODEL / 8;
  cvt_kernel<<<nx8 / 256, 256, 0, stream>>>(x, xh, nx8, 1.0f);
  // fold 1/sqrt(Dh)=1/8 into Wq so attention scores arrive pre-scaled
  cvt_kernel<<<nw8 / 256, 256, 0, stream>>>(Wq, wh, nw8, 0.125f);
  cvt_kernel<<<nw8 / 256, 256, 0, stream>>>(Wk, wh + (size_t)D_MODEL * D_MODEL, nw8, 1.0f);
  cvt_kernel<<<nw8 / 256, 256, 0, stream>>>(Wv, wh + (size_t)2 * D_MODEL * D_MODEL, nw8, 1.0f);

  // 3 matrices x (64 M-tiles x 8 N-tiles) = 1536 blocks of 256 threads
  proj_kernel<<<1536, 256, 0, stream>>>(xh, wh, qh, kh, vt);

  dim3 ag(SEQ / 16, N_HEADS, BATCH);
  attn_kernel<<<ag, 32, 0, stream>>>(qh, kh, vt, out);
}
